// HeadAttn_88802743812907
// MI455X (gfx1250) — matrix operand order repaired and checked by reading, not compiled, not run on hardware
//
#include <hip/hip_runtime.h>
#include <hip/hip_bf16.h>

// ---------------- problem constants ----------------
#define BATCH 32
#define SEQ   256
#define DMODEL 1024
#define NLAYER 4
#define NHEAD 8
#define DKH   128           // DMODEL / NHEAD
#define PASTW 16
#define FUTW  4
#define DIMW  128
#define NOUT  (FUTW * DIMW) // 512
#define MROWS (BATCH * SEQ)            // 8192
#define M2ROWS (BATCH * (SEQ - PASTW)) // 7680

// ---------------- WMMA types (CDNA5, wave32) ----------------
typedef __attribute__((ext_vector_type(16))) __bf16 v16bf;
typedef __attribute__((ext_vector_type(8)))  float  v8f;

union Frag { uint4 u[2]; v16bf v; };

__device__ __forceinline__ unsigned f2bf(float f) {
    unsigned u = __float_as_uint(f);
    unsigned r = u + 0x7FFFu + ((u >> 16) & 1u);   // round-to-nearest-even
    return r >> 16;
}

// ---------------- gfx1250 async global->LDS path ----------------
#if __has_builtin(__builtin_amdgcn_global_load_async_to_lds_b128) && \
    __has_builtin(__builtin_amdgcn_s_wait_asynccnt)
#define USE_ASYNC 1
#else
#define USE_ASYNC 0
#endif

typedef int vint4 __attribute__((vector_size(16)));
typedef __attribute__((address_space(1))) vint4 gvint4;   // global int4
typedef __attribute__((address_space(3))) vint4 lvint4;   // LDS int4

__device__ __forceinline__ void cp_async16(unsigned short* l, const unsigned short* g) {
#if USE_ASYNC
    __builtin_amdgcn_global_load_async_to_lds_b128((gvint4*)(g), (lvint4*)(l), 0, 0);
#else
    *(uint4*)l = *(const uint4*)g;
#endif
}

template <unsigned short N>
__device__ __forceinline__ void cp_wait() {
#if USE_ASYNC
    __builtin_amdgcn_s_wait_asynccnt(N);
#endif
}

constexpr int LDA = 40;   // bf16 elems per LDS row; 80B stride -> conflict-free b128 reads
constexpr int LDB = 40;
constexpr int TILE_HALF = 128 * LDA;   // one double-buffer half (bf16 elems)

// =====================================================================
// GEMM: C[M x N] = A[M x K] (bf16) * Wt[N x K]^T (bf16, pre-transposed)
//       + bias, f32 accumulate via v_wmma_f32_16x16x32_bf16.
// Block = 256 threads = 8 waves; tile 128x128; wave tile 64x32.
// Double-buffered LDS, async global->LDS staging (ASYNCcnt).
// gather!=0: A row r reads source row (r/240)*256 + r%240 + PASTW.
// =====================================================================
__device__ __forceinline__ void stage_tile(const unsigned short* __restrict__ A,
                                           const unsigned short* __restrict__ Wt,
                                           unsigned short* la, unsigned short* lb,
                                           int m0, int n0, int k0, int K,
                                           int gather, int tid)
{
#pragma unroll
    for (int i = 0; i < 2; ++i) {
        int c    = tid + i * 256;       // 16B-chunk id, 0..511
        int row  = c >> 2;              // 0..127
        int part = (c & 3) * 8;         // bf16 offset within 32-wide K slice
        int rowA = m0 + row;
        int srcR = gather ? ((rowA / 240) * 256 + (rowA % 240) + PASTW) : rowA;
        cp_async16(la + row * LDA + part, A  + (size_t)srcR * K + k0 + part);
        cp_async16(lb + row * LDB + part, Wt + (size_t)(n0 + row) * K + k0 + part);
    }
}

__global__ __launch_bounds__(256)
void gemm_bf16_wmma(const unsigned short* __restrict__ A,
                    const unsigned short* __restrict__ Wt,
                    const float* __restrict__ bias, float* __restrict__ C,
                    int N, int K, int gather)
{
    __shared__ __align__(16) unsigned short lds_a[2 * TILE_HALF];
    __shared__ __align__(16) unsigned short lds_b[2 * TILE_HALF];

    const int tid  = threadIdx.x;
    const int lane = tid & 31;
    const int wid  = tid >> 5;
    const int wm   = wid >> 2;   // 0..1  -> 64-row slab
    const int wn   = wid & 3;    // 0..3  -> 32-col slab
    const int n0   = blockIdx.x * 128;
    const int m0   = blockIdx.y * 128;

    // A-fragment lane mapping (16x32 bf16): lanes 0-15 K=0..7/16..23, 16-31 K=8..15/24..31
    const int arow = lane & 15;
    const int akb  = (lane >> 4) << 3;   // 0 or 8
    // B-fragment lane mapping (32x16): lane = column, lanes 0-15 K=0..7/16..23, 16-31 K=8..15/24..31
    const int bcol = lane & 15;
    const int bkb  = (lane >> 4) << 3;   // 0 or 8

    v8f acc[4][2];
#pragma unroll
    for (int mi = 0; mi < 4; ++mi)
#pragma unroll
        for (int ni = 0; ni < 2; ++ni)
            acc[mi][ni] = v8f{0.f, 0.f, 0.f, 0.f, 0.f, 0.f, 0.f, 0.f};

    // prologue: stage first K-tile into buffer 0
    stage_tile(A, Wt, lds_a, lds_b, m0, n0, 0, K, gather, tid);
    int cur = 0;

    for (int k0 = 0; k0 < K; k0 += 32) {
        unsigned short* la = lds_a + cur * TILE_HALF;
        unsigned short* lb = lds_b + cur * TILE_HALF;

        if (k0 + 32 < K) {   // stage next tile into the other buffer, then drain current
            stage_tile(A, Wt, lds_a + (cur ^ 1) * TILE_HALF, lds_b + (cur ^ 1) * TILE_HALF,
                       m0, n0, k0 + 32, K, gather, tid);
            cp_wait<4>();    // 4 just-issued stay outstanding; current tile's 4 complete
        } else {
            cp_wait<0>();
        }
        __syncthreads();     // everyone's current-tile data visible in LDS

        Frag af[4], bf[2];
#pragma unroll
        for (int mi = 0; mi < 4; ++mi) {
            int base = (wm * 64 + mi * 16 + arow) * LDA + akb;
            af[mi].u[0] = *(const uint4*)&la[base];
            af[mi].u[1] = *(const uint4*)&la[base + 16];
        }
#pragma unroll
        for (int ni = 0; ni < 2; ++ni) {
            int base = (wn * 32 + ni * 16 + bcol) * LDB + bkb;
            bf[ni].u[0] = *(const uint4*)&lb[base];
            bf[ni].u[1] = *(const uint4*)&lb[base + 16];
        }
#pragma unroll
        for (int mi = 0; mi < 4; ++mi)
#pragma unroll
            for (int ni = 0; ni < 2; ++ni)
                acc[mi][ni] = __builtin_amdgcn_wmma_f32_16x16x32_bf16(
                    false, af[mi].v, false, bf[ni].v,
                    (short)0, acc[mi][ni], false, false);

        __syncthreads();     // all reads of current buffer done before it is rewritten
        cur ^= 1;
    }

    // ---- epilogue: C/D layout = lanes 0-15 M=r, 16-31 M=r+8, col=lane&15 ----
    const int lrow = (lane >> 4) << 3;
    const int lcol = lane & 15;
#pragma unroll
    for (int mi = 0; mi < 4; ++mi) {
#pragma unroll
        for (int ni = 0; ni < 2; ++ni) {
            int cg = n0 + wn * 32 + ni * 16 + lcol;
            float bv = bias[cg];
#pragma unroll
            for (int r = 0; r < 8; ++r) {
                int rg = m0 + wm * 64 + mi * 16 + lrow + r;
                C[(size_t)rg * N + cg] = acc[mi][ni][r] + bv;
            }
        }
    }
}

// =====================================================================
// Weight convert + transpose: Wt[n][k] (bf16) = W[k][n] (fp32)
// Grid (N/32, K/32), block (32, 8).
// =====================================================================
__global__ __launch_bounds__(256)
void convert_transpose(const float* __restrict__ W, unsigned short* __restrict__ Wt,
                       int K, int N)
{
    __shared__ unsigned short t[32][33];
    const int n0 = blockIdx.x * 32, k0 = blockIdx.y * 32;
    const int tx = threadIdx.x, ty = threadIdx.y;
#pragma unroll
    for (int r = 0; r < 4; ++r) {
        int kk = ty * 4 + r;
        t[kk][tx] = (unsigned short)f2bf(W[(size_t)(k0 + kk) * N + n0 + tx]);
    }
    __syncthreads();
#pragma unroll
    for (int r = 0; r < 4; ++r) {
        int nn = ty * 4 + r;
        Wt[(size_t)(n0 + nn) * K + k0 + tx] = t[tx][nn];
    }
}

// =====================================================================
// x (fp32) and xb (bf16) copies of the input
// =====================================================================
__global__ void cvt_x(const float4* __restrict__ src, float4* __restrict__ x,
                      uint2* __restrict__ xb, int n4)
{
    int i = blockIdx.x * blockDim.x + threadIdx.x;
    if (i < n4) {
        float4 f = src[i];
        x[i] = f;
        uint2 p;
        p.x = f2bf(f.x) | (f2bf(f.y) << 16);
        p.y = f2bf(f.z) | (f2bf(f.w) << 16);
        xb[i] = p;
    }
}

// =====================================================================
// Banded attention: one wave per (b, h, query). Lanes 0..nk-1 own keys.
// Writes bf16 ctx (feeds the Wo GEMM) and the FULL attn row (zeros
// outside the window) into d_out's attn region (doubles as zero-fill).
// =====================================================================
__global__ __launch_bounds__(256)
void attn_window(const float* __restrict__ q, const float* __restrict__ k,
                 const float* __restrict__ v, unsigned short* __restrict__ ctxb,
                 float* __restrict__ attn, int layer)
{
    const int tid  = threadIdx.x;
    const int lane = tid & 31;
    const int wid  = tid >> 5;
    const int gid  = blockIdx.x * 8 + wid;     // 0 .. B*H*L-1
    const int i = gid & (SEQ - 1);
    const int h = (gid >> 8) & (NHEAD - 1);
    const int b = gid >> 11;

    int jlo = i - PASTW; if (jlo < 0) jlo = 0;
    const int nk = i - jlo + 1;                // <= 17

    const size_t qoff = ((size_t)(b * SEQ + i)) * DMODEL + h * DKH;
    float score = -1e30f;
    if (lane < nk) {
        const float* qp = q + qoff;
        const float* kp = k + ((size_t)(b * SEQ + jlo + lane)) * DMODEL + h * DKH;
        float s = 0.f;
#pragma unroll 8
        for (int d = 0; d < DKH; ++d) s = fmaf(qp[d], kp[d], s);
        score = s * 0.08838834764831845f;      // 1/sqrt(128)
    }
    float m = score;
#pragma unroll
    for (int off = 16; off; off >>= 1) m = fmaxf(m, __shfl_xor(m, off, 32));
    float e = (lane < nk) ? __expf(score - m) : 0.f;
    float ssum = e;
#pragma unroll
    for (int off = 16; off; off >>= 1) ssum += __shfl_xor(ssum, off, 32);
    const float a = e / ssum;

    // context: lanes split DK (4 contiguous elems each), emit bf16
    const int d0 = lane * 4;
    float a0 = 0.f, a1 = 0.f, a2 = 0.f, a3 = 0.f;
    for (int j = 0; j < nk; ++j) {
        float aj = __shfl(a, j, 32);
        const float* vp = v + ((size_t)(b * SEQ + jlo + j)) * DMODEL + h * DKH + d0;
        a0 = fmaf(aj, vp[0], a0); a1 = fmaf(aj, vp[1], a1);
        a2 = fmaf(aj, vp[2], a2); a3 = fmaf(aj, vp[3], a3);
    }
    uint2 p;
    p.x = f2bf(a0) | (f2bf(a1) << 16);
    p.y = f2bf(a2) | (f2bf(a3) << 16);
    ((uint2*)ctxb)[(qoff + d0) >> 2] = p;

    // full attention row (B, NL, H, L, L)
    float* arow = attn + ((((size_t)b * NLAYER + layer) * NHEAD + h) * SEQ + i) * (size_t)SEQ;
#pragma unroll
    for (int t = 0; t < 8; ++t) {
        int c = lane * 8 + t;
        int idx = c - jlo;
        bool valid = (idx >= 0) && (idx < nk);
        float av = __shfl(a, valid ? idx : 0, 32);
        arow[c] = valid ? av : 0.f;
    }
}

// =====================================================================
// out = LayerNorm(resid + proj) * g + beta — also emits bf16 copy outb
// =====================================================================
__global__ __launch_bounds__(256)
void add_ln(const float* __restrict__ resid, const float* __restrict__ proj,
            const float* __restrict__ g, const float* __restrict__ beta,
            float* __restrict__ out, unsigned short* __restrict__ outb)
{
    __shared__ float red[256];
    const int r = blockIdx.x;
    const int t = threadIdx.x;
    const size_t base = (size_t)r * DMODEL;

    float x[4];
#pragma unroll
    for (int i = 0; i < 4; ++i) {
        int c = t * 4 + i;
        x[i] = resid[base + c] + proj[base + c];
    }
    float s = x[0] + x[1] + x[2] + x[3];
    red[t] = s; __syncthreads();
    for (int off = 128; off; off >>= 1) {
        if (t < off) red[t] += red[t + off];
        __syncthreads();
    }
    const float mu = red[0] * (1.0f / DMODEL);
    __syncthreads();
    float vs = 0.f;
#pragma unroll
    for (int i = 0; i < 4; ++i) { float d = x[i] - mu; vs = fmaf(d, d, vs); }
    red[t] = vs; __syncthreads();
    for (int off = 128; off; off >>= 1) {
        if (t < off) red[t] += red[t + off];
        __syncthreads();
    }
    const float inv = rsqrtf(red[0] * (1.0f / DMODEL) + 1e-5f);
    float o[4];
#pragma unroll
    for (int i = 0; i < 4; ++i) {
        int c = t * 4 + i;
        o[i] = (x[i] - mu) * inv * g[c] + beta[c];
        out[base + c] = o[i];
    }
    uint2 p;
    p.x = f2bf(o[0]) | (f2bf(o[1]) << 16);
    p.y = f2bf(o[2]) | (f2bf(o[3]) << 16);
    ((uint2*)outb)[(base >> 2) + t] = p;
}

// =====================================================================
extern "C" void kernel_launch(void* const* d_in, const int* in_sizes, int n_in,
                              void* d_out, int out_size, void* d_ws, size_t ws_size,
                              hipStream_t stream)
{
    (void)in_sizes; (void)n_in; (void)out_size; (void)ws_size;
    const float* inp   = (const float*)d_in[0];
    const float* Wq    = (const float*)d_in[1];
    const float* bq    = (const float*)d_in[2];
    const float* Wk    = (const float*)d_in[3];
    const float* bk    = (const float*)d_in[4];
    const float* Wv    = (const float*)d_in[5];
    const float* bv    = (const float*)d_in[6];
    const float* Wo    = (const float*)d_in[7];
    const float* bo    = (const float*)d_in[8];
    const float* ln_g  = (const float*)d_in[9];
    const float* ln_b  = (const float*)d_in[10];
    const float* W_out = (const float*)d_in[11];
    const float* b_out = (const float*)d_in[12];

    float* y    = (float*)d_out;                                  // (B,240,4,128)
    float* attn = y + (size_t)M2ROWS * NOUT;                      // (B,NL,H,L,L)

    const size_t MD = (size_t)MROWS * DMODEL;
    const size_t DD = (size_t)DMODEL * DMODEL;
    float* x    = (float*)d_ws;       // current layer activations (fp32, residual)
    float* q    = x + MD;
    float* kbuf = q + MD;
    float* vbuf = kbuf + MD;
    float* proj = q;                  // reuse q after attention
    unsigned short* xb   = (unsigned short*)(vbuf + MD);  // bf16 activations
    unsigned short* ctxb = xb + MD;                       // bf16 attention context
    unsigned short* wt   = ctxb + MD;                     // bf16 transposed weights
    unsigned short* wtq  = wt;
    unsigned short* wtk  = wt + 4 * DD;
    unsigned short* wtv  = wt + 8 * DD;
    unsigned short* wto  = wt + 12 * DD;
    unsigned short* wtout = wt + 16 * DD;                 // (512 x 1024)

    const dim3 blk(256);
    const dim3 tblk(32, 8);
    const dim3 gT(DMODEL / 32, DMODEL / 32);      // (32, 32) per DxD weight
    const dim3 gTout(NOUT / 32, DMODEL / 32);     // (16, 32) for W_out
    const dim3 gD(DMODEL / 128, MROWS / 128);     // (8, 64)
    const dim3 gOut(NOUT / 128, M2ROWS / 128);    // (4, 60)

    // ---- pre-convert all weights to transposed bf16 (one cheap pass) ----
    for (int l = 0; l < NLAYER; ++l) {
        convert_transpose<<<gT, tblk, 0, stream>>>(Wq + l * DD, wtq + l * DD, DMODEL, DMODEL);
        convert_transpose<<<gT, tblk, 0, stream>>>(Wk + l * DD, wtk + l * DD, DMODEL, DMODEL);
        convert_transpose<<<gT, tblk, 0, stream>>>(Wv + l * DD, wtv + l * DD, DMODEL, DMODEL);
        convert_transpose<<<gT, tblk, 0, stream>>>(Wo + l * DD, wto + l * DD, DMODEL, DMODEL);
    }
    convert_transpose<<<gTout, tblk, 0, stream>>>(W_out, wtout, DMODEL, NOUT);

    // ---- x = inp (fp32 + bf16) ----
    {
        int n4 = (int)(MD / 4);
        cvt_x<<<(n4 + 255) / 256, blk, 0, stream>>>((const float4*)inp, (float4*)x,
                                                    (uint2*)xb, n4);
    }

    for (int l = 0; l < NLAYER; ++l) {
        const size_t wOff = (size_t)l * DD;
        const size_t bOff = (size_t)l * DMODEL;
        gemm_bf16_wmma<<<gD, blk, 0, stream>>>(xb, wtq + wOff, bq + bOff, q,    DMODEL, DMODEL, 0);
        gemm_bf16_wmma<<<gD, blk, 0, stream>>>(xb, wtk + wOff, bk + bOff, kbuf, DMODEL, DMODEL, 0);
        gemm_bf16_wmma<<<gD, blk, 0, stream>>>(xb, wtv + wOff, bv + bOff, vbuf, DMODEL, DMODEL, 0);
        attn_window<<<BATCH * NHEAD * SEQ / 8, blk, 0, stream>>>(q, kbuf, vbuf, ctxb, attn, l);
        gemm_bf16_wmma<<<gD, blk, 0, stream>>>(ctxb, wto + wOff, bo + bOff, proj, DMODEL, DMODEL, 0);
        add_ln<<<MROWS, blk, 0, stream>>>(x, proj, ln_g + bOff, ln_b + bOff, x, xb);
    }

    // y = x[:, PAST:] @ W_out + b_out   (row-gather built into the GEMM)
    gemm_bf16_wmma<<<gOut, blk, 0, stream>>>(xb, wtout, b_out, y, NOUT, DMODEL, 1);
}
